// GraphTransformer_30116310679881
// MI455X (gfx1250) — matrix operand order repaired and checked by reading, not compiled, not run on hardware
//
#include <hip/hip_runtime.h>
#include <hip/hip_bf16.h>
#include <stdint.h>

#define NNODES 20000
#define NEDGES 640000
#define HID    64
#define HEADS  4
#define GRAPHS 512
#define NOUT   128
#define PEDIM  10
#define NLAYER 3

typedef __bf16 v16bf __attribute__((ext_vector_type(16)));
typedef float  v8f   __attribute__((ext_vector_type(8)));
union Frag { v16bf v; int4 q[2]; };

// ---------- helpers ----------
__device__ __forceinline__ unsigned short f2bf(float f) {
  unsigned u = __float_as_uint(f);
  u += 0x7fffu + ((u >> 16) & 1u);
  return (unsigned short)(u >> 16);
}
__device__ __forceinline__ float bf2f(unsigned short h) {
  return __uint_as_float(((unsigned)h) << 16);
}
// order-preserving float<->uint encoding for atomicMax
__device__ __forceinline__ unsigned encF(float x) {
  unsigned u = __float_as_uint(x);
  return (u & 0x80000000u) ? ~u : (u | 0x80000000u);
}
__device__ __forceinline__ float decF(unsigned u) {
  unsigned v = (u & 0x80000000u) ? (u & 0x7fffffffu) : ~u;
  return __uint_as_float(v);
}

// A-fragment (16x32 bf16) load from row-major bf16, row stride 64 halves.
// lane<16:  K = 0..7 , 16..23 ; lane>=16: K = 8..15, 24..31 (CDNA5 ISA A layout)
__device__ __forceinline__ void load_a(Frag& a0, Frag& a1,
                                       const unsigned short* row, int hi) {
  a0.q[0] = *(const int4*)(row + (hi ? 8 : 0));
  a0.q[1] = *(const int4*)(row + (hi ? 24 : 16));
  a1.q[0] = *(const int4*)(row + 32 + (hi ? 8 : 0));
  a1.q[1] = *(const int4*)(row + 32 + (hi ? 24 : 16));
}

// one 16x16 output tile: C = A(16x64) * W(64x16-col-tile), K split into 2 WMMAs
__device__ __forceinline__ v8f gemm_tile(const Frag& a0, const Frag& a1,
                                         const unsigned short* wt, int mat,
                                         int nt, int lane) {
  Frag b0, b1;
  const unsigned short* t0 = wt + ((size_t)mat * 8 + 0 + nt) * 512 + lane * 16;
  const unsigned short* t1 = wt + ((size_t)mat * 8 + 4 + nt) * 512 + lane * 16;
  b0.q[0] = *(const int4*)t0;  b0.q[1] = *(const int4*)(t0 + 8);
  b1.q[0] = *(const int4*)t1;  b1.q[1] = *(const int4*)(t1 + 8);
  v8f c = {0.f,0.f,0.f,0.f,0.f,0.f,0.f,0.f};
  c = __builtin_amdgcn_wmma_f32_16x16x32_bf16(false, a0.v, false, b0.v, (short)0, c, false, false);
  c = __builtin_amdgcn_wmma_f32_16x16x32_bf16(false, a1.v, false, b1.v, (short)0, c, false, false);
  return c;
}

// full 16x64 projection, compile-time-specialized store target (no branches)
__device__ __forceinline__ void proj_bf16(const Frag& a0, const Frag& a1,
                                          const unsigned short* wt, int mat,
                                          const float* bias, int n0, int hi,
                                          int col, int lane,
                                          unsigned short* dst) {
  #pragma unroll
  for (int nt = 0; nt < 4; ++nt) {
    v8f c = gemm_tile(a0, a1, wt, mat, nt, lane);
    float bb = bias[nt * 16 + col];
    #pragma unroll
    for (int r = 0; r < 8; ++r)
      dst[(size_t)(n0 + r + 8 * hi) * HID + nt * 16 + col] = f2bf(c[r] + bb);
  }
}
__device__ __forceinline__ void proj_f32(const Frag& a0, const Frag& a1,
                                         const unsigned short* wt, int mat,
                                         const float* bias, int n0, int hi,
                                         int col, int lane, float* dst) {
  #pragma unroll
  for (int nt = 0; nt < 4; ++nt) {
    v8f c = gemm_tile(a0, a1, wt, mat, nt, lane);
    float bb = bias[nt * 16 + col];
    #pragma unroll
    for (int r = 0; r < 8; ++r)
      dst[(size_t)(n0 + r + 8 * hi) * HID + nt * 16 + col] = c[r] + bb;
  }
}

// ---------- workspace layout (all 256B aligned) ----------
constexpr size_t OFF_WT  = 0;                                    // 136 tiles * 1024B
constexpr size_t OFF_X32 = 139264;                               // f32 x      N*64*4
constexpr size_t OFF_XB  = OFF_X32 + (size_t)NNODES*HID*4;       // bf16 x     N*64*2
constexpr size_t OFF_Q   = OFF_XB  + (size_t)NNODES*HID*2;       // bf16 q
constexpr size_t OFF_K   = OFF_Q   + (size_t)NNODES*HID*2;       // bf16 k
constexpr size_t OFF_V   = OFF_K   + (size_t)NNODES*HID*2;       // bf16 v
constexpr size_t OFF_XS  = OFF_V   + (size_t)NNODES*HID*2;       // f32 x@Ws+bs
constexpr size_t OFF_EE  = OFF_XS  + (size_t)NNODES*HID*4;       // bf16 e_emb E*64*2
constexpr size_t OFF_LG  = OFF_EE  + (size_t)NEDGES*HID*2;       // f32 logits E*4
constexpr size_t OFF_MX  = OFF_LG  + (size_t)NEDGES*HEADS*4;     // u32 max    N*4
constexpr size_t OFF_DEN = OFF_MX  + (size_t)NNODES*HEADS*4;     // f32 den    N*4
constexpr size_t OFF_AGG = OFF_DEN + (size_t)NNODES*HEADS*4;     // f32 agg    N*64
constexpr size_t OFF_PL  = OFF_AGG + (size_t)NNODES*HID*4;       // f32 pooled G*64
constexpr size_t OFF_CT  = OFF_PL  + (size_t)GRAPHS*HID*4;       // f32 counts G
constexpr size_t OFF_PB  = OFF_CT  + (size_t)GRAPHS*4;           // bf16 pooled G*64

// ---------- kernels ----------

// pack all weight matrices into bf16 WMMA B-fragment tiles.
// tile id: mats 0..14 = (layer*5 + {q,k,v,e,s}) * 8 + kf*4 + nt ; fc = 120 + kf*8 + nt
__global__ void k_prep_w(const float* Wq, const float* Wk, const float* Wv,
                         const float* We, const float* Ws, const float* fcW,
                         unsigned short* wt) {
  int tid = blockIdx.x;
  int lane = threadIdx.x;
  const float* src; int cols, kf, nt;
  if (tid < 120) {
    int mat = tid >> 3, sub = tid & 7;
    int layer = mat / 5, which = mat % 5;
    kf = sub >> 2; nt = sub & 3; cols = HID;
    const float* w = (which==0)?Wq:(which==1)?Wk:(which==2)?Wv:(which==3)?We:Ws;
    src = w + (size_t)layer * HID * HID;
  } else {
    int t = tid - 120; kf = t >> 3; nt = t & 7; cols = NOUT; src = fcW;
  }
  unsigned short* dst = wt + (size_t)tid * 512 + lane * 16;
  int n = nt * 16 + (lane & 15);
  int kbase = kf * 32 + ((lane >> 4) ? 8 : 0);
  #pragma unroll
  for (int h = 0; h < 16; ++h)
    dst[h] = f2bf(src[(size_t)(kbase + (h < 8 ? h : h + 8)) * cols + n]);
}

// atom encoder + SignNet + RWSE MLPs. one block (64 threads) per node.
__global__ __launch_bounds__(64) void k_node_enc(
    const int* x_atom, const float* lap, const float* rw, const float* aemb,
    const float* sW1, const float* sb1, const float* sW2, const float* sb2,
    const float* rW1, const float* rb1, const float* rW2, const float* rb2,
    float* x32, unsigned short* xb) {
  __shared__ float hp[HID], hn[HID], hr[HID];
  int n = blockIdx.x, j = threadIdx.x;
  float ap = sb1[j], an = sb1[j], ar = rb1[j];
  for (int p = 0; p < PEDIM; ++p) {
    float lp = lap[n * PEDIM + p];
    float w  = sW1[p * HID + j];
    ap += lp * w; an -= lp * w;
    ar += rw[n * PEDIM + p] * rW1[p * HID + j];
  }
  hp[j] = fmaxf(ap, 0.f); hn[j] = fmaxf(an, 0.f); hr[j] = fmaxf(ar, 0.f);
  __syncthreads();
  float acc = 2.f * sb2[j] + rb2[j];
  for (int f = 0; f < 9; ++f) {
    int idx = x_atom[n * 9 + f];
    acc += aemb[((size_t)f * 119 + idx) * HID + j];
  }
  for (int k = 0; k < HID; ++k)
    acc += (hp[k] + hn[k]) * sW2[k * HID + j] + hr[k] * rW2[k * HID + j];
  x32[(size_t)n * HID + j] = acc;
  xb[(size_t)n * HID + j]  = f2bf(acc);
}

// bond encoder -> bf16 e_emb. thread = (edge, 2 columns).
__global__ void k_bond_enc(const int* eattr, const float* bemb,
                           unsigned short* ee) {
  int id = blockIdx.x * blockDim.x + threadIdx.x;
  if (id >= NEDGES * 32) return;
  int m = id >> 5, c = (id & 31) * 2;
  int a0 = eattr[m * 3], a1 = eattr[m * 3 + 1], a2 = eattr[m * 3 + 2];
  const float* t0 = bemb + (size_t)a0 * HID;
  const float* t1 = bemb + (size_t)(5 + a1) * HID;
  const float* t2 = bemb + (size_t)(10 + a2) * HID;
  float v0 = t0[c] + t1[c] + t2[c];
  float v1 = t0[c + 1] + t1[c + 1] + t2[c + 1];
  unsigned pk = (unsigned)f2bf(v0) | ((unsigned)f2bf(v1) << 16);
  *(unsigned*)(ee + (size_t)m * HID + c) = pk;
}

// per-layer scratch init
__global__ void k_zero(float* agg, float* den, unsigned* mx) {
  int id = blockIdx.x * blockDim.x + threadIdx.x;
  if (id < NNODES * HID) agg[id] = 0.f;
  if (id < NNODES * HEADS) { den[id] = 0.f; mx[id] = 0x007fffffu; /* enc(-inf) */ }
}

// q/k/v and x@Ws projections via WMMA. one wave = 16 nodes.
__global__ __launch_bounds__(256) void k_node_gemm(
    const unsigned short* xb, const unsigned short* wt,
    const float* bq, const float* bk, const float* bv, const float* bs,
    int layer, unsigned short* q, unsigned short* k, unsigned short* v,
    float* xs) {
  int wave = (blockIdx.x * blockDim.x + threadIdx.x) >> 5;
  int lane = threadIdx.x & 31;
  int n0 = wave * 16;
  if (n0 >= NNODES) return;
  int hi = lane >> 4, col = lane & 15;
  Frag a0, a1;
  load_a(a0, a1, xb + (size_t)(n0 + col) * HID, hi);
  int mbase = layer * 5;
  proj_bf16(a0, a1, wt, mbase + 0, bq + (size_t)layer * HID, n0, hi, col, lane, q);
  proj_bf16(a0, a1, wt, mbase + 1, bk + (size_t)layer * HID, n0, hi, col, lane, k);
  proj_bf16(a0, a1, wt, mbase + 2, bv + (size_t)layer * HID, n0, hi, col, lane, v);
  proj_f32 (a0, a1, wt, mbase + 4, bs + (size_t)layer * HID, n0, hi, col, lane, xs);
}

// edge pass 1: e = e_emb@We (WMMA); logits = q[dst].(k[src]+e)/4 ; atomicMax per (dst,head)
__global__ __launch_bounds__(256) void k_edge_p1(
    const unsigned short* ee, const unsigned short* wt,
    const unsigned short* qb, const unsigned short* kb,
    const int* eidx, const float* be, int layer,
    float* logits, unsigned* mx) {
  int wave = (blockIdx.x * blockDim.x + threadIdx.x) >> 5;
  int lane = threadIdx.x & 31;
  int m0 = wave * 16;
  if (m0 >= NEDGES) return;
  int hi = lane >> 4, col = lane & 15;
  Frag a0, a1;
  load_a(a0, a1, ee + (size_t)(m0 + col) * HID, hi);
  int srcs[8], dsts[8];
  #pragma unroll
  for (int r = 0; r < 8; ++r) {
    int m = m0 + r + 8 * hi;
    srcs[r] = eidx[m];
    dsts[r] = eidx[NEDGES + m];
  }
  int mat = layer * 5 + 3;
  float lg[4][8];
  #pragma unroll
  for (int t = 0; t < 4; ++t) {
    v8f c = gemm_tile(a0, a1, wt, mat, t, lane);
    float bb = be[(size_t)layer * HID + t * 16 + col];
    #pragma unroll
    for (int r = 0; r < 8; ++r) {
      float ev = c[r] + bb;
      float kv = bf2f(kb[(size_t)srcs[r] * HID + t * 16 + col]);
      float qv = bf2f(qb[(size_t)dsts[r] * HID + t * 16 + col]);
      lg[t][r] = qv * (ev + kv);
    }
  }
  #pragma unroll
  for (int t = 0; t < 4; ++t)
    #pragma unroll
    for (int r = 0; r < 8; ++r) {
      float s = lg[t][r];
      s += __shfl_xor(s, 1, 16);
      s += __shfl_xor(s, 2, 16);
      s += __shfl_xor(s, 4, 16);
      s += __shfl_xor(s, 8, 16);
      lg[t][r] = s * 0.25f;          // 1/sqrt(HEAD_C)
    }
  if (col == 0) {
    #pragma unroll
    for (int t = 0; t < 4; ++t)
      #pragma unroll
      for (int r = 0; r < 8; ++r) {
        int m = m0 + r + 8 * hi;
        logits[(size_t)m * 4 + t] = lg[t][r];
        atomicMax(&mx[(size_t)dsts[r] * 4 + t], encF(lg[t][r]));
      }
  }
}

// edge pass 2: recompute e via WMMA; scatter numerator and (deferred) denominator.
__global__ __launch_bounds__(256) void k_edge_p2(
    const unsigned short* ee, const unsigned short* wt,
    const unsigned short* vb, const int* eidx, const float* be, int layer,
    const float* logits, const unsigned* mx, float* agg, float* den) {
  int wave = (blockIdx.x * blockDim.x + threadIdx.x) >> 5;
  int lane = threadIdx.x & 31;
  int m0 = wave * 16;
  if (m0 >= NEDGES) return;
  int hi = lane >> 4, col = lane & 15;
  Frag a0, a1;
  load_a(a0, a1, ee + (size_t)(m0 + col) * HID, hi);
  int srcs[8], dsts[8];
  #pragma unroll
  for (int r = 0; r < 8; ++r) {
    int m = m0 + r + 8 * hi;
    srcs[r] = eidx[m];
    dsts[r] = eidx[NEDGES + m];
  }
  int mat = layer * 5 + 3;
  float wv[4][8];                    // softmax weights (uniform across col group)
  #pragma unroll
  for (int t = 0; t < 4; ++t) {
    v8f c = gemm_tile(a0, a1, wt, mat, t, lane);
    float bb = be[(size_t)layer * HID + t * 16 + col];
    #pragma unroll
    for (int r = 0; r < 8; ++r) {
      int m = m0 + r + 8 * hi;
      float w = __expf(logits[(size_t)m * 4 + t] -
                       decF(mx[(size_t)dsts[r] * 4 + t]));
      wv[t][r] = w;
      float ev = c[r] + bb;
      float vvv = bf2f(vb[(size_t)srcs[r] * HID + t * 16 + col]);
      atomicAdd(&agg[(size_t)dsts[r] * HID + t * 16 + col], w * (ev + vvv));
    }
  }
  if (col == 0) {                    // one lane per 16-lane group: denominator
    #pragma unroll
    for (int t = 0; t < 4; ++t)
      #pragma unroll
      for (int r = 0; r < 8; ++r)
        atomicAdd(&den[(size_t)dsts[r] * 4 + t], wv[t][r]);
  }
}

// x = relu(agg/den + x@Ws + bs)
__global__ void k_node_upd(const float* agg, const float* den, const float* xs,
                           float* x32, unsigned short* xb) {
  int id = blockIdx.x * blockDim.x + threadIdx.x;
  if (id >= NNODES * HID) return;
  int n = id >> 6, h = (id & 63) >> 4;
  float val = agg[id] / (den[n * 4 + h] + 1e-16f) + xs[id];
  val = fmaxf(val, 0.f);
  x32[id] = val;
  xb[id]  = f2bf(val);
}

__global__ void k_pool_zero(float* pooled, float* cnt) {
  int id = blockIdx.x * blockDim.x + threadIdx.x;
  if (id < GRAPHS * HID) pooled[id] = 0.f;
  if (id < GRAPHS) cnt[id] = 0.f;
}
__global__ void k_pool(const float* x32, const int* batch, float* pooled,
                       float* cnt) {
  int id = blockIdx.x * blockDim.x + threadIdx.x;
  if (id >= NNODES * HID) return;
  int n = id >> 6, c = id & 63;
  int g = batch[n];
  atomicAdd(&pooled[(size_t)g * HID + c], x32[id]);
  if (c == 0) atomicAdd(&cnt[g], 1.0f);
}
__global__ void k_pool_norm(const float* pooled, const float* cnt,
                            unsigned short* pb) {
  int id = blockIdx.x * blockDim.x + threadIdx.x;
  if (id >= GRAPHS * HID) return;
  pb[id] = f2bf(pooled[id] / fmaxf(cnt[id >> 6], 1.0f));
}

// final FC: (512x64) @ (64x128) via WMMA, fp32 out
__global__ __launch_bounds__(256) void k_fc(const unsigned short* pb,
                                            const unsigned short* wt,
                                            const float* fcb, float* out) {
  int wave = (blockIdx.x * blockDim.x + threadIdx.x) >> 5;
  int lane = threadIdx.x & 31;
  int r0 = wave * 16;
  if (r0 >= GRAPHS) return;
  int hi = lane >> 4, col = lane & 15;
  Frag a0, a1;
  load_a(a0, a1, pb + (size_t)(r0 + col) * HID, hi);
  #pragma unroll
  for (int nt = 0; nt < 8; ++nt) {
    Frag b0, b1;
    const unsigned short* t0 = wt + (size_t)(120 + nt) * 512 + lane * 16;
    const unsigned short* t1 = wt + (size_t)(128 + nt) * 512 + lane * 16;
    b0.q[0] = *(const int4*)t0;  b0.q[1] = *(const int4*)(t0 + 8);
    b1.q[0] = *(const int4*)t1;  b1.q[1] = *(const int4*)(t1 + 8);
    v8f c = {0.f,0.f,0.f,0.f,0.f,0.f,0.f,0.f};
    c = __builtin_amdgcn_wmma_f32_16x16x32_bf16(false, a0.v, false, b0.v, (short)0, c, false, false);
    c = __builtin_amdgcn_wmma_f32_16x16x32_bf16(false, a1.v, false, b1.v, (short)0, c, false, false);
    float bb = fcb[nt * 16 + col];
    #pragma unroll
    for (int r = 0; r < 8; ++r)
      out[(size_t)(r0 + r + 8 * hi) * NOUT + nt * 16 + col] = c[r] + bb;
  }
}

// ---------- launch ----------
extern "C" void kernel_launch(void* const* d_in, const int* in_sizes, int n_in,
                              void* d_out, int out_size, void* d_ws,
                              size_t ws_size, hipStream_t stream) {
  const int*   x_atom = (const int*)d_in[0];
  const int*   eidx   = (const int*)d_in[1];
  const int*   eattr  = (const int*)d_in[2];
  const int*   batch  = (const int*)d_in[3];
  const float* lap    = (const float*)d_in[4];
  const float* rwse   = (const float*)d_in[5];
  const float* aemb   = (const float*)d_in[6];
  const float* bemb   = (const float*)d_in[7];
  const float* sW1 = (const float*)d_in[8];  const float* sb1 = (const float*)d_in[9];
  const float* sW2 = (const float*)d_in[10]; const float* sb2 = (const float*)d_in[11];
  const float* rW1 = (const float*)d_in[12]; const float* rb1 = (const float*)d_in[13];
  const float* rW2 = (const float*)d_in[14]; const float* rb2 = (const float*)d_in[15];
  const float* Wq = (const float*)d_in[16];  const float* bq = (const float*)d_in[17];
  const float* Wk = (const float*)d_in[18];  const float* bk = (const float*)d_in[19];
  const float* Wv = (const float*)d_in[20];  const float* bv = (const float*)d_in[21];
  const float* We = (const float*)d_in[22];  const float* be = (const float*)d_in[23];
  const float* Ws = (const float*)d_in[24];  const float* bs = (const float*)d_in[25];
  const float* fcW = (const float*)d_in[26]; const float* fcb = (const float*)d_in[27];

  char* ws = (char*)d_ws;
  unsigned short* wt  = (unsigned short*)(ws + OFF_WT);
  float*          x32 = (float*)(ws + OFF_X32);
  unsigned short* xb  = (unsigned short*)(ws + OFF_XB);
  unsigned short* qb  = (unsigned short*)(ws + OFF_Q);
  unsigned short* kb  = (unsigned short*)(ws + OFF_K);
  unsigned short* vb  = (unsigned short*)(ws + OFF_V);
  float*          xs  = (float*)(ws + OFF_XS);
  unsigned short* ee  = (unsigned short*)(ws + OFF_EE);
  float*          lgb = (float*)(ws + OFF_LG);
  unsigned*       mx  = (unsigned*)(ws + OFF_MX);
  float*          den = (float*)(ws + OFF_DEN);
  float*          agg = (float*)(ws + OFF_AGG);
  float*          pl  = (float*)(ws + OFF_PL);
  float*          ct  = (float*)(ws + OFF_CT);
  unsigned short* pb  = (unsigned short*)(ws + OFF_PB);
  float*          out = (float*)d_out;

  hipLaunchKernelGGL(k_prep_w, dim3(136), dim3(32), 0, stream,
                     Wq, Wk, Wv, We, Ws, fcW, wt);
  hipLaunchKernelGGL(k_node_enc, dim3(NNODES), dim3(64), 0, stream,
                     x_atom, lap, rwse, aemb, sW1, sb1, sW2, sb2,
                     rW1, rb1, rW2, rb2, x32, xb);
  hipLaunchKernelGGL(k_bond_enc, dim3((NEDGES * 32 + 255) / 256), dim3(256), 0,
                     stream, eattr, bemb, ee);

  for (int l = 0; l < NLAYER; ++l) {
    hipLaunchKernelGGL(k_zero, dim3((NNODES * HID + 255) / 256), dim3(256), 0,
                       stream, agg, den, mx);
    hipLaunchKernelGGL(k_node_gemm, dim3((NNODES / 16 + 7) / 8), dim3(256), 0,
                       stream, xb, wt, bq, bk, bv, bs, l, qb, kb, vb, xs);
    hipLaunchKernelGGL(k_edge_p1, dim3(NEDGES / 16 / 8), dim3(256), 0, stream,
                       ee, wt, qb, kb, eidx, be, l, lgb, mx);
    hipLaunchKernelGGL(k_edge_p2, dim3(NEDGES / 16 / 8), dim3(256), 0, stream,
                       ee, wt, vb, eidx, be, l, lgb, mx, agg, den);
    hipLaunchKernelGGL(k_node_upd, dim3((NNODES * HID + 255) / 256), dim3(256),
                       0, stream, agg, den, xs, x32, xb);
  }

  hipLaunchKernelGGL(k_pool_zero, dim3((GRAPHS * HID + 255) / 256), dim3(256),
                     0, stream, pl, ct);
  hipLaunchKernelGGL(k_pool, dim3((NNODES * HID + 255) / 256), dim3(256), 0,
                     stream, x32, batch, pl, ct);
  hipLaunchKernelGGL(k_pool_norm, dim3((GRAPHS * HID + 255) / 256), dim3(256),
                     0, stream, pl, ct, pb);
  hipLaunchKernelGGL(k_fc, dim3((GRAPHS / 16 + 7) / 8), dim3(256), 0, stream,
                     pb, wt, fcb, out);
}
